// TransformerDecoderBlock_19559281066517
// MI455X (gfx1250) — compile-verified, matrix operand order checked by reading, not run on hardware
//
#include <hip/hip_runtime.h>
#include <hip/hip_bf16.h>
#include <math.h>

// ---------------------------------------------------------------------------
// Types for CDNA5 WMMA (wave32): v_wmma_f32_16x16x32_f16
// ---------------------------------------------------------------------------
typedef _Float16 half8 __attribute__((ext_vector_type(8)));
typedef _Float16 v16h  __attribute__((ext_vector_type(16)));
typedef float    v8f   __attribute__((ext_vector_type(8)));

#define WMMA_F16(Afrag, Bfrag, Cacc) \
  __builtin_amdgcn_wmma_f32_16x16x32_f16(false, (Afrag), false, (Bfrag), (short)0, (Cacc), false, false)

// Load one 16x32 f16 A/B fragment for v_wmma_f32_16x16x32_f16.
// ISA layout: lanes 0-15 -> line (lane), K = {0..7, 16..23};
//             lanes 16-31 -> line (lane-16), K = {8..15, 24..31}.
__device__ __forceinline__ v16h frag_ld(const _Float16* __restrict__ base,
                                        size_t line0, size_t stride, int e0) {
  const int lane = threadIdx.x & 31;
  const _Float16* p = base + (line0 + (size_t)(lane & 15)) * stride
                           + (size_t)(e0 + ((lane & 16) ? 8 : 0));
  half8 lo = *(const half8*)(p);
  half8 hi = *(const half8*)(p + 16);
  return __builtin_shufflevector(lo, hi, 0,1,2,3,4,5,6,7,8,9,10,11,12,13,14,15);
}

// ---------------------------------------------------------------------------
// Elementwise f32 -> f16 convert
// ---------------------------------------------------------------------------
__global__ void cvt_f32_to_f16(const float* __restrict__ in,
                               _Float16* __restrict__ out, long long n) {
  long long i = (long long)blockIdx.x * blockDim.x + threadIdx.x;
  if (i < n) out[i] = (_Float16)in[i];
}

// Transpose + convert: Wt[n][k] = (f16) W[k][n].  K, N multiples of 16.
__global__ void transpose_cvt(const float* __restrict__ W,
                              _Float16* __restrict__ Wt, int K, int N) {
  int k = blockIdx.x * 16 + threadIdx.x;
  int n = blockIdx.y * 16 + threadIdx.y;
  Wt[(size_t)n * K + k] = (_Float16)W[(size_t)k * N + n];
}

// ---------------------------------------------------------------------------
// GEMM:  C[M,N] = A[M,K] (f16, row-major) * Wt[N,K]^T (f16) + bias
// Workgroup tile 128x128, 8 waves, wave tile 32x64 (2x4 WMMA subtiles).
// relu: apply max(0,.)   storeT: write out16 as [B, N, Sdim] (V^T layout)
// ---------------------------------------------------------------------------
__global__ void __launch_bounds__(256, 1)
gemm_f16_wmma(const _Float16* __restrict__ A,
              const _Float16* __restrict__ Wt,
              const float* __restrict__ bias,
              _Float16* __restrict__ out16,
              float* __restrict__ out32,
              int M, int N, int K, int relu, int storeT, int Sdim) {
  const int lane = threadIdx.x & 31;
  const int w    = threadIdx.x >> 5;
  const int m0   = blockIdx.y * 128 + (w >> 1) * 32;
  const int n0   = blockIdx.x * 128 + (w & 1) * 64;
  const int hi8  = (lane & 16) ? 8 : 0;

  // Per-lane base pointers for fragment lines (offsets become immediates).
  const _Float16* pa0 = A  + (size_t)(m0 + (lane & 15)) * K + ((lane & 16) ? 8 : 0);
  const _Float16* pb0 = Wt + (size_t)(n0 + (lane & 15)) * K + ((lane & 16) ? 8 : 0);

  v8f acc[2][4];
#pragma unroll
  for (int i = 0; i < 2; ++i)
#pragma unroll
    for (int j = 0; j < 4; ++j) acc[i][j] = {};

  for (int e0 = 0; e0 < K; e0 += 32) {
    // speculative prefetch of the next K-step (dropped silently if OOB)
    __builtin_prefetch(pa0 + e0 + 32);
    __builtin_prefetch(pb0 + e0 + 32);
    v16h a[2], b[4];
#pragma unroll
    for (int i = 0; i < 2; ++i) a[i] = frag_ld(A,  (size_t)(m0 + i * 16), (size_t)K, e0);
#pragma unroll
    for (int j = 0; j < 4; ++j) b[j] = frag_ld(Wt, (size_t)(n0 + j * 16), (size_t)K, e0);
#pragma unroll
    for (int i = 0; i < 2; ++i)
#pragma unroll
      for (int j = 0; j < 4; ++j)
        acc[i][j] = WMMA_F16(a[i], b[j], acc[i][j]);
  }

#pragma unroll
  for (int i = 0; i < 2; ++i)
#pragma unroll
    for (int j = 0; j < 4; ++j)
#pragma unroll
      for (int a2 = 0; a2 < 8; ++a2) {
        int m = m0 + i * 16 + a2 + hi8;
        int n = n0 + j * 16 + (lane & 15);
        float v = acc[i][j][a2] + bias[n];
        if (relu) v = fmaxf(v, 0.0f);
        if (storeT) {
          int bb = m / Sdim, s = m - bb * Sdim;
          out16[((size_t)bb * N + n) * Sdim + s] = (_Float16)v;
        } else {
          size_t idx = (size_t)m * N + n;
          if (out16) out16[idx] = (_Float16)v;
          if (out32) out32[idx] = v;
        }
      }
}

// ---------------------------------------------------------------------------
// Flash attention (single head, dim E).  Q,K: f16 [B,S,E]; VT: f16 [B,E,S].
// Workgroup: 32 query rows, loops over 64-key tiles, online softmax.
// Each wave keeps 32(q) x 96(e) of O in registers (12 x v8f).
// ---------------------------------------------------------------------------
__global__ void __launch_bounds__(256, 1)
attn_wmma(const _Float16* __restrict__ Q,
          const _Float16* __restrict__ Kh,
          const _Float16* __restrict__ VT,
          float* __restrict__ Out,
          int S, int E, int causal, float scale) {
  __shared__ __align__(16) float    Ss[32 * 64];
  __shared__ __align__(16) _Float16 Ps[32 * 64];
  __shared__ float mrow[32], lrow[32], frow[32];

  const int tid  = threadIdx.x;
  const int lane = tid & 31;
  const int w    = tid >> 5;
  const int hi8  = (lane & 16) ? 8 : 0;
  const int q0   = blockIdx.x * 32;
  const int b    = blockIdx.y;

  const _Float16* Qb = Q  + (size_t)b * S * E;
  const _Float16* Kb = Kh + (size_t)b * S * E;
  const _Float16* Vb = VT + (size_t)b * E * S;

  v8f O[2][6];
#pragma unroll
  for (int mi = 0; mi < 2; ++mi)
#pragma unroll
    for (int nn = 0; nn < 6; ++nn) O[mi][nn] = {};

  if (tid < 32) { mrow[tid] = -3.0e38f; lrow[tid] = 0.0f; }
  __syncthreads();

  const int nkt = causal ? (q0 / 64 + 1) : (S / 64);
  for (int kt = 0; kt < nkt; ++kt) {
    const int kt0 = kt * 64;

    // ---- phase 1: scores S = scale * Q Ktile^T  (wave w -> one 16x16 tile)
    {
      const int mi = w >> 2, ni = w & 3;
      v8f sacc = {};
      for (int e0 = 0; e0 < E; e0 += 32) {
        v16h aq = frag_ld(Qb, (size_t)(q0  + mi * 16), (size_t)E, e0);
        v16h bk = frag_ld(Kb, (size_t)(kt0 + ni * 16), (size_t)E, e0);
        sacc = WMMA_F16(aq, bk, sacc);
      }
#pragma unroll
      for (int a2 = 0; a2 < 8; ++a2) {
        int qr = mi * 16 + a2 + hi8;
        int kc = ni * 16 + (lane & 15);
        float v = sacc[a2] * scale;
        if (causal && (kt0 + kc) > (q0 + qr)) v = -1.0e30f;
        Ss[qr * 64 + kc] = v;
      }
    }
    __syncthreads();

    // ---- phase 2: online softmax row update.
    // 8 lanes per query row, shfl_xor reductions (width 8).
    {
      const int r  = tid >> 3;        // 0..31
      const int cb = (tid & 7) * 8;   // 8 columns per lane
      float mx = -3.0e38f;
#pragma unroll
      for (int c = 0; c < 8; ++c) mx = fmaxf(mx, Ss[r * 64 + cb + c]);
#pragma unroll
      for (int offd = 4; offd > 0; offd >>= 1) mx = fmaxf(mx, __shfl_xor(mx, offd, 8));
      float mold = mrow[r];
      float mnew = fmaxf(mold, mx);
      float sum = 0.0f;
#pragma unroll
      for (int c = 0; c < 8; ++c) {
        float p = __expf(Ss[r * 64 + cb + c] - mnew);
        Ps[r * 64 + cb + c] = (_Float16)p;
        sum += p;
      }
#pragma unroll
      for (int offd = 4; offd > 0; offd >>= 1) sum += __shfl_xor(sum, offd, 8);
      if ((tid & 7) == 0) {
        float f = __expf(mold - mnew);
        mrow[r] = mnew;
        lrow[r] = lrow[r] * f + sum;
        frow[r] = f;
      }
    }
    __syncthreads();

    // ---- phase 3: rescale O and accumulate P * Vtile
#pragma unroll
    for (int mi = 0; mi < 2; ++mi)
#pragma unroll
      for (int a2 = 0; a2 < 8; ++a2) {
        float f = frow[mi * 16 + a2 + hi8];
#pragma unroll
        for (int nn = 0; nn < 6; ++nn) O[mi][nn][a2] *= f;
      }

    for (int kk = 0; kk < 64; kk += 32) {
      v16h pa[2];
#pragma unroll
      for (int mi = 0; mi < 2; ++mi)
        pa[mi] = frag_ld((const _Float16*)Ps, (size_t)(mi * 16), 64, kk);
#pragma unroll
      for (int nn = 0; nn < 6; ++nn) {
        v16h bv = frag_ld(Vb, (size_t)(w * 96 + nn * 16), (size_t)S, kt0 + kk);
#pragma unroll
        for (int mi = 0; mi < 2; ++mi)
          O[mi][nn] = WMMA_F16(pa[mi], bv, O[mi][nn]);
      }
    }
  }
  __syncthreads();

  // ---- epilogue: O / l -> Out (f32)
#pragma unroll
  for (int mi = 0; mi < 2; ++mi)
#pragma unroll
    for (int nn = 0; nn < 6; ++nn)
#pragma unroll
      for (int a2 = 0; a2 < 8; ++a2) {
        int qr  = mi * 16 + a2 + hi8;
        int col = w * 96 + nn * 16 + (lane & 15);
        Out[((size_t)b * S + q0 + qr) * E + col] = O[mi][nn][a2] / lrow[qr];
      }
}

// ---------------------------------------------------------------------------
// y = LayerNorm(x + h) * g + b ; writes f32 (residual chain) and f16 (GEMM in)
// One 256-thread block per row of E=768.
// ---------------------------------------------------------------------------
__global__ void __launch_bounds__(256, 1)
add_ln(const float* __restrict__ xb, const float* __restrict__ hb,
       const float* __restrict__ g, const float* __restrict__ bb,
       float* __restrict__ y32, _Float16* __restrict__ y16, int E) {
  const int row = blockIdx.x;
  const int tid = threadIdx.x;
  __shared__ float buf[768];
  __shared__ float red[256];

  const float* xp = xb + (size_t)row * E;
  const float* hp = hb + (size_t)row * E;

  float s = 0.0f;
  for (int e = tid; e < E; e += 256) {
    float v = xp[e] + hp[e];
    buf[e] = v;
    s += v;
  }
  red[tid] = s;
  __syncthreads();
  for (int off = 128; off > 0; off >>= 1) {
    if (tid < off) red[tid] += red[tid + off];
    __syncthreads();
  }
  float mu = red[0] / (float)E;
  __syncthreads();

  float s2 = 0.0f;
  for (int e = tid; e < E; e += 256) {
    float d = buf[e] - mu;
    s2 += d * d;
  }
  red[tid] = s2;
  __syncthreads();
  for (int off = 128; off > 0; off >>= 1) {
    if (tid < off) red[tid] += red[tid + off];
    __syncthreads();
  }
  float rs = rsqrtf(red[0] / (float)E + 1e-5f);

  for (int e = tid; e < E; e += 256) {
    float y = (buf[e] - mu) * rs * g[e] + bb[e];
    size_t idx = (size_t)row * E + e;
    if (y32) y32[idx] = y;
    if (y16) y16[idx] = (_Float16)y;
  }
}

// ---------------------------------------------------------------------------
// Host orchestration
// ---------------------------------------------------------------------------
extern "C" void kernel_launch(void* const* d_in, const int* in_sizes, int n_in,
                              void* d_out, int out_size, void* d_ws, size_t ws_size,
                              hipStream_t stream) {
  (void)in_sizes; (void)n_in; (void)out_size; (void)ws_size;
  constexpr int B = 8, S = 2048, E = 768, H = 3072;
  constexpr int M = B * S;
  const float scale = 1.0f / sqrtf((float)E);

  const float* x      = (const float*)d_in[0];
  const float* kv     = (const float*)d_in[1];
  const float* wq_w   = (const float*)d_in[2];   const float* wq_b   = (const float*)d_in[3];
  const float* wk_w   = (const float*)d_in[4];   const float* wk_b   = (const float*)d_in[5];
  const float* wv_w   = (const float*)d_in[6];   const float* wv_b   = (const float*)d_in[7];
  const float* ln1_g  = (const float*)d_in[8];   const float* ln1_b  = (const float*)d_in[9];
  const float* wq2_w  = (const float*)d_in[10];  const float* wq2_b  = (const float*)d_in[11];
  const float* wk2_w  = (const float*)d_in[12];  const float* wk2_b  = (const float*)d_in[13];
  const float* wv2_w  = (const float*)d_in[14];  const float* wv2_b  = (const float*)d_in[15];
  const float* ln2_g  = (const float*)d_in[16];  const float* ln2_b  = (const float*)d_in[17];
  const float* mlp_w1 = (const float*)d_in[18];  const float* mlp_b1 = (const float*)d_in[19];
  const float* mlp_w2 = (const float*)d_in[20];  const float* mlp_b2 = (const float*)d_in[21];
  const float* ln3_g  = (const float*)d_in[22];  const float* ln3_b  = (const float*)d_in[23];
  float* out = (float*)d_out;

  char* ws = (char*)d_ws;
  size_t off = 0;
  auto alloc = [&](size_t bytes) -> void* {
    void* p = (void*)(ws + off);
    off += (bytes + 255) & ~(size_t)255;
    return p;
  };

  const size_t nME = (size_t)M * E;
  _Float16* xh    = (_Float16*)alloc(nME * 2);
  _Float16* kvh   = (_Float16*)alloc(nME * 2);
  _Float16* wqT   = (_Float16*)alloc((size_t)E * E * 2);
  _Float16* wkT   = (_Float16*)alloc((size_t)E * E * 2);
  _Float16* wvT   = (_Float16*)alloc((size_t)E * E * 2);
  _Float16* wq2T  = (_Float16*)alloc((size_t)E * E * 2);
  _Float16* wk2T  = (_Float16*)alloc((size_t)E * E * 2);
  _Float16* wv2T  = (_Float16*)alloc((size_t)E * E * 2);
  _Float16* w1T   = (_Float16*)alloc((size_t)E * H * 2);
  _Float16* w2T   = (_Float16*)alloc((size_t)H * E * 2);
  _Float16* qb    = (_Float16*)alloc(nME * 2);
  _Float16* kb    = (_Float16*)alloc(nME * 2);
  _Float16* vTb   = (_Float16*)alloc(nME * 2);   // [B,E,S]
  float*    h1    = (float*)   alloc(nME * 4);   // attn / mlp f32 output (reused)
  float*    y32   = (float*)   alloc(nME * 4);
  _Float16* y16   = (_Float16*)alloc(nME * 2);
  float*    x2_32 = (float*)   alloc(nME * 4);
  _Float16* x2_16 = (_Float16*)alloc(nME * 2);
  _Float16* hmid  = (_Float16*)alloc((size_t)M * H * 2);

  // --- convert activations and weights to f16 (weights transposed) ---
  cvt_f32_to_f16<<<(unsigned)(nME / 256), 256, 0, stream>>>(x,  xh,  (long long)nME);
  cvt_f32_to_f16<<<(unsigned)(nME / 256), 256, 0, stream>>>(kv, kvh, (long long)nME);
  dim3 tb(16, 16);
  transpose_cvt<<<dim3(E / 16, E / 16), tb, 0, stream>>>(wq_w,  wqT,  E, E);
  transpose_cvt<<<dim3(E / 16, E / 16), tb, 0, stream>>>(wk_w,  wkT,  E, E);
  transpose_cvt<<<dim3(E / 16, E / 16), tb, 0, stream>>>(wv_w,  wvT,  E, E);
  transpose_cvt<<<dim3(E / 16, E / 16), tb, 0, stream>>>(wq2_w, wq2T, E, E);
  transpose_cvt<<<dim3(E / 16, E / 16), tb, 0, stream>>>(wk2_w, wk2T, E, E);
  transpose_cvt<<<dim3(E / 16, E / 16), tb, 0, stream>>>(wv2_w, wv2T, E, E);
  transpose_cvt<<<dim3(E / 16, H / 16), tb, 0, stream>>>(mlp_w1, w1T, E, H);
  transpose_cvt<<<dim3(H / 16, E / 16), tb, 0, stream>>>(mlp_w2, w2T, H, E);

  // --- self attention projections ---
  dim3 gE(E / 128, M / 128);
  gemm_f16_wmma<<<gE, 256, 0, stream>>>(xh, wqT, wq_b, qb,  nullptr, M, E, E, 0, 0, S);
  gemm_f16_wmma<<<gE, 256, 0, stream>>>(xh, wkT, wk_b, kb,  nullptr, M, E, E, 0, 0, S);
  gemm_f16_wmma<<<gE, 256, 0, stream>>>(xh, wvT, wv_b, vTb, nullptr, M, E, E, 0, 1, S);

  // --- causal self attention ---
  attn_wmma<<<dim3(S / 32, B), 256, 0, stream>>>(qb, kb, vTb, h1, S, E, 1, scale);
  add_ln<<<M, 256, 0, stream>>>(x, h1, ln1_g, ln1_b, y32, y16, E);

  // --- cross attention ---
  gemm_f16_wmma<<<gE, 256, 0, stream>>>(y16, wq2T, wq2_b, qb,  nullptr, M, E, E, 0, 0, S);
  gemm_f16_wmma<<<gE, 256, 0, stream>>>(kvh, wk2T, wk2_b, kb,  nullptr, M, E, E, 0, 0, S);
  gemm_f16_wmma<<<gE, 256, 0, stream>>>(kvh, wv2T, wv2_b, vTb, nullptr, M, E, E, 0, 1, S);
  attn_wmma<<<dim3(S / 32, B), 256, 0, stream>>>(qb, kb, vTb, h1, S, E, 0, scale);
  add_ln<<<M, 256, 0, stream>>>(y32, h1, ln2_g, ln2_b, x2_32, x2_16, E);

  // --- MLP ---
  gemm_f16_wmma<<<dim3(H / 128, M / 128), 256, 0, stream>>>(x2_16, w1T, mlp_b1, hmid, nullptr, M, H, E, 1, 0, S);
  gemm_f16_wmma<<<gE, 256, 0, stream>>>(hmid, w2T, mlp_b2, nullptr, h1, M, E, H, 1, 0, S);
  add_ln<<<M, 256, 0, stream>>>(x2_32, h1, ln3_g, ln3_b, out, nullptr, E);
}
